// ContextQueryAttention_80573586473274
// MI455X (gfx1250) — compile-verified, matrix operand order checked by reading, not run on hardware
//
#include <hip/hip_runtime.h>

#define DEVI __device__ __forceinline__

typedef __bf16 v16bf __attribute__((ext_vector_type(16)));
typedef __bf16 v8bf  __attribute__((ext_vector_type(8)));
typedef __bf16 v4bf  __attribute__((ext_vector_type(4)));
typedef float  v8f   __attribute__((ext_vector_type(8)));
typedef float  v4f   __attribute__((ext_vector_type(4)));

static constexpr int NB  = 256;
static constexpr int LC  = 400;
static constexpr int LQ  = 50;
static constexpr int D   = 128;
static constexpr int LQP = 64;   // LQ padded to multiple of 32 (WMMA K) / 16 (N)
static constexpr int LCP = 416;  // LC padded to multiple of 32 (K of t-GEMM)
static constexpr int SROW = 53;  // odd LDS row stride -> conflict-free column access

// ---------------- workspace layout (bytes) ----------------
static constexpr size_t OFF_XCW  = 0;                                  // bf16 (NB,LC,D)   x_cont * W2
static constexpr size_t OFF_XCT  = OFF_XCW  + (size_t)NB*LC*D*2;       // bf16 (NB,D,LCP)  x_cont^T, K-padded
static constexpr size_t OFF_XQT  = OFF_XCT  + (size_t)NB*D*LCP*2;      // bf16 (NB,D,LQP)  x_ques^T, K-padded
static constexpr size_t OFF_XQ   = OFF_XQT  + (size_t)NB*D*LQP*2;      // bf16 (NB,LQP,D)  x_ques, row-padded
static constexpr size_t OFF_SC   = OFF_XQ   + (size_t)NB*LQP*D*2;      // f32  (NB,LC)     x_cont . W0
static constexpr size_t OFF_SQ   = OFF_SC   + (size_t)NB*LC*4;         // f32  (NB,LQ)     x_ques . W1
static constexpr size_t OFF_SBAR = OFF_SQ   + (size_t)NB*LQ*4;         // bf16 (NB,LC,LQP) S_bar, zero-padded
static constexpr size_t OFF_ST   = OFF_SBAR + (size_t)NB*LC*LQP*2;     // bf16 (NB,LQP,LCP) S_T, zero-padded
static constexpr size_t OFF_TT   = OFF_ST   + (size_t)NB*LQP*LCP*2;    // bf16 (NB,D,LQP)  (S_T @ x_cont)^T

// ---------------- output layout (floats) ----------------
static constexpr size_t OUT_SBAR = (size_t)NB*LC*4*D;                  // after result (B,LC,512)
static constexpr size_t OUT_ST   = OUT_SBAR + (size_t)NB*LC*LQ;

// ---------------- wave helpers (wave32) ----------------
DEVI float wsum(float v) { for (int o = 16; o; o >>= 1) v += __shfl_xor(v, o, 32); return v; }
DEVI float wmax_red(float v) { for (int o = 16; o; o >>= 1) v = fmaxf(v, __shfl_xor(v, o, 32)); return v; }

// Load a 16-element bf16 WMMA fragment slice: elements 0..7 at p, 8..15 at p+16,
// matching the ISA 16-bit A/B layout (lane<16: K=kb..kb+7 & kb+16..kb+23, lane>=16: +8).
DEVI v16bf frag_from(const __bf16* p) {
    v8bf lo = *(const v8bf*)p;
    v8bf hi = *(const v8bf*)(p + 16);
    return __builtin_shufflevector(lo, hi, 0,1,2,3,4,5,6,7,8,9,10,11,12,13,14,15);
}

// ======================= kernel 1: context prep =======================
__global__ void prep_cont(const float* __restrict__ xc, const float* __restrict__ W0,
                          const float* __restrict__ W2, __bf16* __restrict__ xcw,
                          __bf16* __restrict__ xcT, float* __restrict__ s_c) {
    int wave = blockIdx.x * 8 + (threadIdx.x >> 5);
    int lane = threadIdx.x & 31;
    int b = wave / LCP, c = wave % LCP;
    if (b >= NB) return;
    int d0 = lane * 4;
    if (c < LC) {
        v4f v  = *(const v4f*)(xc + ((size_t)(b * LC + c)) * D + d0);
        v4f w2 = *(const v4f*)(W2 + d0);
        v4f w0 = *(const v4f*)(W0 + d0);
        v4bf cw = { (__bf16)(v[0] * w2[0]), (__bf16)(v[1] * w2[1]),
                    (__bf16)(v[2] * w2[2]), (__bf16)(v[3] * w2[3]) };
        *(v4bf*)(xcw + ((size_t)(b * LC + c)) * D + d0) = cw;
        for (int j = 0; j < 4; ++j)
            xcT[((size_t)b * D + d0 + j) * LCP + c] = (__bf16)v[j];
        float part = v[0]*w0[0] + v[1]*w0[1] + v[2]*w0[2] + v[3]*w0[3];
        part = wsum(part);
        if (lane == 0) s_c[b * LC + c] = part;
    } else {  // K-padding columns of xcT
        for (int j = 0; j < 4; ++j)
            xcT[((size_t)b * D + d0 + j) * LCP + c] = (__bf16)0.f;
    }
}

// ======================= kernel 2: query prep =======================
__global__ void prep_ques(const float* __restrict__ xq, const float* __restrict__ W1,
                          __bf16* __restrict__ xqb, __bf16* __restrict__ xqT,
                          float* __restrict__ s_q) {
    int wave = blockIdx.x * 8 + (threadIdx.x >> 5);
    int lane = threadIdx.x & 31;
    int b = wave / LQP, q = wave % LQP;
    if (b >= NB) return;
    int d0 = lane * 4;
    if (q < LQ) {
        v4f v  = *(const v4f*)(xq + ((size_t)(b * LQ + q)) * D + d0);
        v4f w1 = *(const v4f*)(W1 + d0);
        v4bf bv = { (__bf16)v[0], (__bf16)v[1], (__bf16)v[2], (__bf16)v[3] };
        *(v4bf*)(xqb + ((size_t)(b * LQP + q)) * D + d0) = bv;
        for (int j = 0; j < 4; ++j)
            xqT[((size_t)b * D + d0 + j) * LQP + q] = (__bf16)v[j];
        float part = v[0]*w1[0] + v[1]*w1[1] + v[2]*w1[2] + v[3]*w1[3];
        part = wsum(part);
        if (lane == 0) s_q[b * LQ + q] = part;
    } else {  // zero-pad rows [50,64) of xqb and K-pad columns of xqT
        v4bf z = { (__bf16)0.f, (__bf16)0.f, (__bf16)0.f, (__bf16)0.f };
        *(v4bf*)(xqb + ((size_t)(b * LQP + q)) * D + d0) = z;
        for (int j = 0; j < 4; ++j)
            xqT[((size_t)b * D + d0 + j) * LQP + q] = (__bf16)0.f;
    }
}

// ========== kernel 3: S = (xc*W2)@xq^T + biases, dual masked softmax ==========
// One workgroup per batch; full S (400x50 f32) lives in LDS (~85 KB of the 320 KB WGP LDS).
__global__ void scores_softmax(const __bf16* __restrict__ xcw, const __bf16* __restrict__ xqb,
                               const float* __restrict__ s_c, const float* __restrict__ s_q,
                               const int* __restrict__ clenv, const int* __restrict__ qlenv,
                               __bf16* __restrict__ wsSbar, __bf16* __restrict__ wsST,
                               float* __restrict__ outSbar, float* __restrict__ outST) {
    extern __shared__ float S[];  // [LC][SROW]
    int b    = blockIdx.x;
    int tid  = threadIdx.x;
    int wave = tid >> 5, lane = tid & 31;
    int kbase = (lane < 16) ? 0 : 8;
    int rr    = lane & 15;
    int half  = (lane < 16) ? 0 : 8;

    // zero the padded S_T bf16 block for this batch (rows q>=50, cols c>=400)
    __bf16* stb = wsST + (size_t)b * LQP * LCP;
    for (int i = tid; i < LQP * LCP; i += 512) stb[i] = (__bf16)0.f;

    const __bf16* A0 = xcw + (size_t)b * LC * D;
    const __bf16* B0 = xqb + (size_t)b * LQP * D;   // row-padded: rows [50,64) are zero

    // --- phase A: 25 x 4 WMMA output tiles of S, 16 waves round-robin ---
    for (int t = wave; t < 100; t += 16) {
        int ct = t >> 2, qt = t & 3;
        const __bf16* arow = A0 + (size_t)(ct * 16 + rr) * D;
        const __bf16* brow = B0 + (size_t)(qt * 16 + rr) * D;
        v8f acc = {};
        for (int ks = 0; ks < 4; ++ks) {
            int k0 = ks * 32 + kbase;
            v16bf af = frag_from(arow + k0);
            v16bf bf = frag_from(brow + k0);
            acc = __builtin_amdgcn_wmma_f32_16x16x32_bf16(false, af, false, bf,
                                                          (short)0, acc, false, false);
        }
        for (int v = 0; v < 8; ++v) {
            int c = ct * 16 + v + half;
            int q = qt * 16 + rr;
            if (q < LQ)
                S[c * SROW + q] = acc[v] + s_c[b * LC + c] + s_q[b * LQ + q];
        }
    }
    __syncthreads();

    int qlen = qlenv[b], clen = clenv[b];

    // --- phase B: row softmax over q -> S_bar ---
    for (int c = wave; c < LC; c += 16) {
        int q1 = lane, q2 = lane + 32;
        float v1 = (q1 < qlen) ? S[c * SROW + q1] : -3.0e38f;
        float v2 = (q2 < qlen) ? S[c * SROW + q2] : -3.0e38f;
        float m  = wmax_red(fmaxf(v1, v2));
        float e1 = (q1 < qlen) ? __expf(v1 - m) : 0.f;
        float e2 = (q2 < qlen) ? __expf(v2 - m) : 0.f;
        float inv = 1.f / wsum(e1 + e2);
        float p1 = e1 * inv, p2 = e2 * inv;
        size_t ro = (size_t)(b * LC + c);
        outSbar[ro * LQ + q1] = p1;
        if (q2 < LQ) outSbar[ro * LQ + q2] = p2;
        wsSbar[ro * LQP + q1] = (__bf16)p1;
        wsSbar[ro * LQP + q2] = (__bf16)p2;   // also zeroes pad q in [50,64)
    }

    // --- phase C: column softmax over c -> S_T (stored (LQ,LC)) ---
    for (int q = wave; q < LQ; q += 16) {
        float m = -3.0e38f;
        for (int c = lane; c < LC; c += 32)
            if (c < clen) m = fmaxf(m, S[c * SROW + q]);
        m = wmax_red(m);
        float s = 0.f;
        for (int c = lane; c < LC; c += 32)
            if (c < clen) s += __expf(S[c * SROW + q] - m);
        float inv = 1.f / wsum(s);
        for (int c = lane; c < LC; c += 32) {
            float p = (c < clen) ? __expf(S[c * SROW + q] - m) * inv : 0.f;
            outST[((size_t)(b * LQ + q)) * LC + c] = p;
            stb[q * LCP + c] = (__bf16)p;
        }
    }
}

// ============== kernel 4: t = S_T @ x_cont, stored transposed bf16 ==============
__global__ void gemm_t(const __bf16* __restrict__ wsST, const __bf16* __restrict__ xcT,
                       __bf16* __restrict__ tT) {
    int b    = blockIdx.x;
    int wave = threadIdx.x >> 5, lane = threadIdx.x & 31;
    int kbase = (lane < 16) ? 0 : 8;
    int rr    = lane & 15;
    int half  = (lane < 16) ? 0 : 8;
    const __bf16* A0 = wsST + (size_t)b * LQP * LCP;
    const __bf16* B0 = xcT  + (size_t)b * D * LCP;
    __bf16* T0 = tT + (size_t)b * D * LQP;
    for (int t = wave; t < 32; t += 8) {
        int qt = t >> 3, dt = t & 7;
        const __bf16* arow = A0 + (size_t)(qt * 16 + rr) * LCP;
        const __bf16* brow = B0 + (size_t)(dt * 16 + rr) * LCP;
        v8f acc = {};
        for (int ks = 0; ks < 13; ++ks) {       // K = 416 (LC padded, tail cols are zero)
            int k0 = ks * 32 + kbase;
            v16bf af = frag_from(arow + k0);
            v16bf bf = frag_from(brow + k0);
            acc = __builtin_amdgcn_wmma_f32_16x16x32_bf16(false, af, false, bf,
                                                          (short)0, acc, false, false);
        }
        for (int v = 0; v < 8; ++v) {
            int q = qt * 16 + v + half;
            int d = dt * 16 + rr;
            T0[(size_t)d * LQP + q] = (__bf16)acc[v];  // q>=50 rows come out zero
        }
    }
}

// ==== kernel 5: c2q = S_bar@xq, q2c = S_bar@t, fused concat epilogue ====
__global__ void gemm_out(const __bf16* __restrict__ wsSbar, const __bf16* __restrict__ xqT,
                         const __bf16* __restrict__ tT, const float* __restrict__ xc,
                         float* __restrict__ out) {
    int b    = blockIdx.x;
    int wave = threadIdx.x >> 5, lane = threadIdx.x & 31;
    int kbase = (lane < 16) ? 0 : 8;
    int rr    = lane & 15;
    int half  = (lane < 16) ? 0 : 8;
    const __bf16* A0 = wsSbar + (size_t)b * LC * LQP;
    const __bf16* B1 = xqT + (size_t)b * D * LQP;
    const __bf16* B2 = tT  + (size_t)b * D * LQP;
    for (int t = wave; t < 200; t += 8) {       // 25 x 8 output tiles
        int ct = t >> 3, dt = t & 7;
        const __bf16* arow  = A0 + (size_t)(ct * 16 + rr) * LQP;
        const __bf16* b1row = B1 + (size_t)(dt * 16 + rr) * LQP;
        const __bf16* b2row = B2 + (size_t)(dt * 16 + rr) * LQP;
        v8f acc1 = {}, acc2 = {};
        for (int ks = 0; ks < 2; ++ks) {        // K = 64 (LQ padded, pad cols are zero)
            int k0 = ks * 32 + kbase;
            v16bf af  = frag_from(arow + k0);
            v16bf bf1 = frag_from(b1row + k0);
            v16bf bf2 = frag_from(b2row + k0);
            acc1 = __builtin_amdgcn_wmma_f32_16x16x32_bf16(false, af, false, bf1,
                                                           (short)0, acc1, false, false);
            acc2 = __builtin_amdgcn_wmma_f32_16x16x32_bf16(false, af, false, bf2,
                                                           (short)0, acc2, false, false);
        }
        for (int v = 0; v < 8; ++v) {
            int c = ct * 16 + v + half;
            int d = dt * 16 + rr;
            float xcv = xc[((size_t)(b * LC + c)) * D + d];
            float c2q = acc1[v], q2c = acc2[v];
            float* o = out + ((size_t)(b * LC + c)) * (4 * D);
            o[d]         = xcv;
            o[D + d]     = c2q;
            o[2 * D + d] = xcv * c2q;
            o[3 * D + d] = xcv * q2c;
        }
    }
}

// ======================= host launch =======================
extern "C" void kernel_launch(void* const* d_in, const int* in_sizes, int n_in,
                              void* d_out, int out_size, void* d_ws, size_t ws_size,
                              hipStream_t stream) {
    (void)in_sizes; (void)n_in; (void)out_size; (void)ws_size;
    const float* xc   = (const float*)d_in[0];
    const float* xq   = (const float*)d_in[1];
    const float* W0   = (const float*)d_in[2];
    const float* W1   = (const float*)d_in[3];
    const float* W2   = (const float*)d_in[4];
    const int*   clen = (const int*)d_in[5];
    const int*   qlen = (const int*)d_in[6];

    char* ws = (char*)d_ws;
    __bf16* xcw    = (__bf16*)(ws + OFF_XCW);
    __bf16* xcT    = (__bf16*)(ws + OFF_XCT);
    __bf16* xqT    = (__bf16*)(ws + OFF_XQT);
    __bf16* xqb    = (__bf16*)(ws + OFF_XQ);
    float*  s_c    = (float*)(ws + OFF_SC);
    float*  s_q    = (float*)(ws + OFF_SQ);
    __bf16* wsSbar = (__bf16*)(ws + OFF_SBAR);
    __bf16* wsST   = (__bf16*)(ws + OFF_ST);
    __bf16* tT     = (__bf16*)(ws + OFF_TT);

    float* out = (float*)d_out;

    prep_cont<<<NB * LCP / 8, 256, 0, stream>>>(xc, W0, W2, xcw, xcT, s_c);
    prep_ques<<<NB * LQP / 8, 256, 0, stream>>>(xq, W1, xqb, xqT, s_q);
    scores_softmax<<<NB, 512, LC * SROW * 4, stream>>>(xcw, xqb, s_c, s_q, clen, qlen,
                                                       wsSbar, wsST,
                                                       out + OUT_SBAR, out + OUT_ST);
    gemm_t<<<NB, 256, 0, stream>>>(wsST, xcT, tT);
    gemm_out<<<NB, 256, 0, stream>>>(wsSbar, xqT, tT, xc, out);
}
